// QuantConv1d_88845693485350
// MI455X (gfx1250) — matrix operand order repaired and checked by reading, not compiled, not run on hardware
//
#include <hip/hip_runtime.h>

// ---------------------------------------------------------------------------
// QuantConv1d (GPT-2 Conv1D, int8 weight): y = x @ W.f32 * scale + bias
//   x [256,4096] f32, W [4096,16384] int8, scale/bias [16384] f32
//   y [256,16384] f32
// CDNA5 wave32 f16-WMMA GEMM, f32 accumulate, LDS double-buffered pipeline.
// BLOCK_M=128 amortizes int8->f16 conversion; perm/magic trick converts
// 4 weight bytes with 1 xor + 2 v_perm_b32 + 2 v_pk_add_f16 (exact in f16).
// ---------------------------------------------------------------------------

typedef __attribute__((ext_vector_type(16))) _Float16 v16h;
typedef __attribute__((ext_vector_type(2)))  _Float16 h2;
typedef __attribute__((ext_vector_type(8)))  float    v8f;
typedef __attribute__((ext_vector_type(4)))  float    v4f;
typedef __attribute__((ext_vector_type(4)))  int      v4i;

#define K_DIM   4096
#define N_DIM   16384
#define M_DIM   256

#define BLOCK_M 128
#define BLOCK_N 128
#define K_CHUNK 32
#define NCHUNK  (K_DIM / K_CHUNK)   // 128

#define A_HALFS (8 * 32 * 16)       // 8 m-tiles * 32 lanes * 16 halfs = 4096
#define B_HALFS (8 * 32 * 16)       // 8 n-tiles * 32 lanes * 16 halfs = 4096
#define BUF_HALFS (A_HALFS + B_HALFS)

union V16U { v16h v; h2 p[8]; unsigned u[8]; };
union H2U  { unsigned u; h2 h; };

// Two packed "0x64 | byte" halfs -> two exact f16 values of the signed bytes.
static __device__ __forceinline__ h2 magic_to_h2(unsigned packed) {
    H2U cv; cv.u = packed;
    const h2 bias1152 = { (_Float16)1152.0f, (_Float16)1152.0f };
    return cv.h - bias1152;   // v_pk_add_f16, exact: (1024 + (c+128)) - 1152 = c
}

__global__ __launch_bounds__(256)
void qconv1d_wmma_kernel(const float* __restrict__ x,
                         const signed char* __restrict__ w,
                         const float* __restrict__ scale,
                         const float* __restrict__ bias,
                         float* __restrict__ out)
{
    __shared__ __align__(32) _Float16 lds[2 * BUF_HALFS];   // 32 KB

    const int tid    = threadIdx.x;
    const int lane   = tid & 31;
    const int wave   = tid >> 5;     // 0..7
    const int wave_m = wave >> 2;    // 0..1 -> m-tiles [4*wm, 4*wm+3]
    const int wave_n = wave & 3;     // 0..3 -> n-tiles [2*wn, 2*wn+1]

    const int n0 = blockIdx.x * BLOCK_N;   // 128 n-blocks
    const int m0 = blockIdx.y * BLOCK_M;   // 2 m-blocks

    // ---- A staging: one fragment-lane per thread (8 mtiles * 32 lanes) ----
    // ISA 16-bit A 16x32: lane<16 : e0..7->K 0..7,  e8..15->K 16..23
    //                     lane>=16: e0..7->K 8..15, e8..15->K 24..31
    const int a_mtile = tid >> 5;              // 0..7
    const int a_lane  = tid & 31;
    const int a_kb    = (a_lane >> 4) << 3;    // 0 or 8
    const int a_row   = m0 + a_mtile * 16 + (a_lane & 15);
    const float* a_gptr = x + (size_t)a_row * K_DIM + a_kb;
    const int a_lds   = (a_mtile * 32 + a_lane) * 16;

    // ---- B staging: (K row, n-tile) per thread ----
    // LDS B 32x16 per n-tile: row = K (b_k), 16 halfs = N 0..15 of the tile.
    const int b_k     = tid & 31;
    const int b_ntile = tid >> 5;              // 0..7
    const signed char* b_gptr = w + (size_t)b_k * N_DIM + n0 + b_ntile * 16;
    const int b_lds   = (b_ntile * 32 + b_k) * 16;

    // accumulators: 4 m-tiles x 2 n-tiles of 16x16 f32 (64 VGPRs)
    v8f acc[4][2];
#pragma unroll
    for (int i = 0; i < 4; ++i)
#pragma unroll
        for (int j = 0; j < 2; ++j)
#pragma unroll
            for (int r = 0; r < 8; ++r) acc[i][j][r] = 0.0f;

    // register prefetch staging
    v4f ar0, ar1, ar2, ar3;   // x: K = kb..kb+7 and kb+16..kb+23
    v4i breg;                 // W: 16 int8

    auto load_chunk = [&](int kc) {
        const float* ap = a_gptr + kc * K_CHUNK;
        ar0 = *(const v4f*)(ap);
        ar1 = *(const v4f*)(ap + 4);
        ar2 = *(const v4f*)(ap + 16);
        ar3 = *(const v4f*)(ap + 20);
        breg = *(const v4i*)(b_gptr + (size_t)kc * K_CHUNK * N_DIM);
    };

    auto store_chunk = [&](int buf) {
        _Float16* A = lds + buf * BUF_HALFS;
        _Float16* B = A + A_HALFS;

        v16h ah;
#pragma unroll
        for (int i = 0; i < 4; ++i) ah[i]      = (_Float16)ar0[i];
#pragma unroll
        for (int i = 0; i < 4; ++i) ah[4 + i]  = (_Float16)ar1[i];
#pragma unroll
        for (int i = 0; i < 4; ++i) ah[8 + i]  = (_Float16)ar2[i];
#pragma unroll
        for (int i = 0; i < 4; ++i) ah[12 + i] = (_Float16)ar3[i];
        *(v16h*)(A + a_lds) = ah;

        V16U bv;
#pragma unroll
        for (int v = 0; v < 4; ++v) {
            const unsigned wx = ((unsigned)breg[v]) ^ 0x80808080u;   // c -> c+128
            // byte splice: [wx.b, 0x64] pairs -> f16 bits of 1024+(c+128)
            const unsigned h01 = __builtin_amdgcn_perm(0x64646464u, wx, 0x04010400u);
            const unsigned h23 = __builtin_amdgcn_perm(0x64646464u, wx, 0x04030402u);
            bv.p[2 * v]     = magic_to_h2(h01);
            bv.p[2 * v + 1] = magic_to_h2(h23);
        }
        *(v16h*)(B + b_lds) = bv.v;
    };

    auto compute = [&](int buf) {
        const _Float16* A = lds + buf * BUF_HALFS;
        const _Float16* B = A + A_HALFS;
        v16h af[4], bf[2];
#pragma unroll
        for (int i = 0; i < 4; ++i) {
            const int mt = wave_m * 4 + i;
            af[i] = *(const v16h*)(A + (mt * 32 + lane) * 16);
        }
#pragma unroll
        for (int j = 0; j < 2; ++j) {
            const int nt = wave_n * 2 + j;
            for (int s = 0; s < 16; ++s) {
                const int k = (s < 8) ? (((lane >> 4) << 3) + s) : (16 + ((lane >> 4) << 3) + (s - 8));
                bf[j][s] = B[(nt * 32 + k) * 16 + (lane & 15)];
            }
        }
#pragma unroll
        for (int i = 0; i < 4; ++i)
#pragma unroll
            for (int j = 0; j < 2; ++j)
                acc[i][j] = __builtin_amdgcn_wmma_f32_16x16x32_f16(
                    false, af[i], false, bf[j],
                    (short)0, acc[i][j], false, false);
    };

    // ---- pipelined main loop: one barrier per K chunk ----
    load_chunk(0);
    store_chunk(0);
    __syncthreads();

    for (int kc = 0; kc < NCHUNK; ++kc) {
        if (kc + 1 < NCHUNK) load_chunk(kc + 1);               // global -> regs
        if (kc + 6 < NCHUNK)                                   // warm L2 ahead
            __builtin_prefetch(b_gptr + (size_t)(kc + 6) * K_CHUNK * N_DIM, 0, 3);
        compute(kc & 1);                                       // ds_load + wmma
        if (kc + 1 < NCHUNK) store_chunk((kc + 1) & 1);        // regs -> other buf
        __syncthreads();
    }

    // ---- epilogue: per-channel scale + bias, f32 store ----
    // C/D layout: VGPR r -> M = r (lanes 0-15) / r+8 (lanes 16-31); N = lane&15.
#pragma unroll
    for (int j = 0; j < 2; ++j) {
        const int col = n0 + (wave_n * 2 + j) * 16 + (lane & 15);
        const float s  = scale[col];
        const float bb = bias[col];
#pragma unroll
        for (int i = 0; i < 4; ++i) {
            const int rbase = m0 + (wave_m * 4 + i) * 16 + ((lane >> 4) << 3);
#pragma unroll
            for (int r = 0; r < 8; ++r) {
                out[(size_t)(rbase + r) * N_DIM + col] = acc[i][j][r] * s + bb;
            }
        }
    }
}

extern "C" void kernel_launch(void* const* d_in, const int* in_sizes, int n_in,
                              void* d_out, int out_size, void* d_ws, size_t ws_size,
                              hipStream_t stream) {
    (void)in_sizes; (void)n_in; (void)out_size; (void)d_ws; (void)ws_size;
    const float*       x     = (const float*)d_in[0];
    const signed char* w     = (const signed char*)d_in[1];   // int8 weight
    const float*       scale = (const float*)d_in[2];
    const float*       bias  = (const float*)d_in[3];
    float*             out   = (float*)d_out;

    dim3 grid(N_DIM / BLOCK_N, M_DIM / BLOCK_M);  // (128, 2)
    dim3 block(256);                              // 8 wave32
    qconv1d_wmma_kernel<<<grid, block, 0, stream>>>(x, w, scale, bias, out);
}
